// PG_Caps_Cifar10_70875550318611
// MI455X (gfx1250) — matrix operand order repaired and checked by reading, not compiled, not run on hardware
//
#include <hip/hip_runtime.h>

typedef __attribute__((ext_vector_type(16))) _Float16 v16h;
typedef __attribute__((ext_vector_type(8)))  float    v8f;

#define EPSBN 1e-5f
#define EPSLN 1e-5f
#define EPSSQ 1e-5f

// ---------------- param prep: fold BN into per-channel scale/shift ----------------
__global__ void prep_ss_kernel(const float* __restrict__ bias,
                               const float* __restrict__ gamma,
                               const float* __restrict__ beta,
                               const float* __restrict__ mu,
                               const float* __restrict__ var,
                               float* __restrict__ scale,
                               float* __restrict__ shift,
                               int C, int has_bn) {
  int c = blockIdx.x * blockDim.x + threadIdx.x;
  if (c >= C) return;
  if (has_bn) {
    float inv = gamma[c] * rsqrtf(var[c] + EPSBN);
    scale[c] = inv;
    shift[c] = bias[c] * inv + beta[c] - mu[c] * inv;
  } else {
    scale[c] = 1.0f;
    shift[c] = bias[c];
  }
}

// W f32 [Cout][Cin][k][k]  ->  Wt f16 [(tap*Cin + ci)][Cout]   (tap = ky*k+kx)
__global__ void prep_wt_kernel(const float* __restrict__ W, _Float16* __restrict__ Wt,
                               int Cout, int Cin, int ksz) {
  int kk2 = ksz * ksz;
  int total = Cout * Cin * kk2;
  int idx = blockIdx.x * blockDim.x + threadIdx.x;
  if (idx >= total) return;
  int tap = idx % kk2;
  int ci  = (idx / kk2) % Cin;
  int co  = idx / (kk2 * Cin);
  Wt[((size_t)tap * Cin + ci) * Cout + co] = (_Float16)W[idx];
}

// ---------------- conv1: 3->64, k=5, pad=2, stride s  (direct, NCHW f32 -> NHWC f16) ----------------
__global__ void conv1_kernel(const float* __restrict__ X, const float* __restrict__ W,
                             const float* __restrict__ scale, const float* __restrict__ shift,
                             _Float16* __restrict__ Y,
                             int B, int Hi, int Wi, int Ho, int Wo, int stride) {
  int idx = blockIdx.x * blockDim.x + threadIdx.x;
  int total = B * Ho * Wo * 64;
  if (idx >= total) return;
  int co = idx & 63;
  int n  = idx >> 6;
  int wo = n % Wo; n /= Wo;
  int ho = n % Ho; int b = n / Ho;
  float acc = 0.f;
  for (int ci = 0; ci < 3; ++ci) {
    for (int ky = 0; ky < 5; ++ky) {
      int yi = ho * stride + ky - 2;
      if (yi < 0 || yi >= Hi) continue;
      for (int kx = 0; kx < 5; ++kx) {
        int xi = wo * stride + kx - 2;
        if (xi < 0 || xi >= Wi) continue;
        acc += X[((b * 3 + ci) * Hi + yi) * Wi + xi] * W[((co * 3 + ci) * 5 + ky) * 5 + kx];
      }
    }
  }
  float v = acc * scale[co] + shift[co];
  v = fmaxf(v, 0.f);
  Y[(((size_t)b * Ho + ho) * Wo + wo) * 64 + co] = (_Float16)v;
}

// ---------------- implicit-GEMM conv with WMMA f16 -> f32 accum ----------------
// X NHWC f16 [B,Hi,Wi,Cin]; Wt f16 [k][Cout] with k = tap*Cin+cin; Y NHWC f16.
// Block = waves*32 threads; tile = (waves*16 couts) x (32 spatial positions).
// Each wave: one A fragment (16 couts x 32 K), two B fragments -> 2 WMMAs per K-chunk.
__global__ void conv_wmma_kernel(const _Float16* __restrict__ X,
                                 const _Float16* __restrict__ Wt,
                                 const float* __restrict__ scale,
                                 const float* __restrict__ shift,
                                 const _Float16* __restrict__ resid,
                                 _Float16* __restrict__ Y,
                                 int B, int Hi, int Wi, int Cin, int cinShift, int Cout,
                                 int Ho, int Wo, int ksz, int stride, int pad, int do_relu) {
  __shared__ __attribute__((aligned(16))) _Float16 a_s[32][136]; // [k-local][cout-local]
  __shared__ __attribute__((aligned(16))) _Float16 b_s[32][40];  // [spatial pos][k-local]
  __shared__ int pos_b[32], pos_y[32], pos_x[32];

  const int tid   = threadIdx.x;
  const int lane  = tid & 31;
  const int wave  = tid >> 5;
  const int nthr  = blockDim.x;        // 128 or 256
  const int waves = nthr >> 5;         // 4 or 8
  const int rowW  = waves * 16;        // A-tile row width (couts per block)
  const int rsh   = (waves == 8) ? 4 : 3;   // log2(uint4 chunks per A row)

  const int n_base = blockIdx.x * 32;
  const int m_base = blockIdx.y * rowW;

  if (tid < 32) {
    int n = n_base + tid;
    int w = n % Wo; n /= Wo;
    int h = n % Ho; int b = n / Ho;
    pos_b[tid] = b; pos_y[tid] = h; pos_x[tid] = w;
  }

  const int row  = lane & 15;
  const int hsel = lane >> 4;          // 0: lanes 0-15, 1: lanes 16-31
  const int mstart = m_base + wave * 16 + hsel * 8;   // 8 contiguous couts per lane

  // hoist loop-invariant epilogue params
  float scl[8], sft[8];
#pragma unroll
  for (int r = 0; r < 8; ++r) { scl[r] = scale[mstart + r]; sft[r] = shift[mstart + r]; }

  union { v8f v; float f[8]; } acc0, acc1;
#pragma unroll
  for (int i = 0; i < 8; ++i) { acc0.f[i] = 0.f; acc1.f[i] = 0.f; }

  const int Ktot   = ksz * ksz * Cin;  // Cin is a power of two (64/128/256)
  const int chunks = Ktot >> 5;
  const int cmask  = Cin - 1;

  for (int ch = 0; ch < chunks; ++ch) {
    const int kk  = ch << 5;
    const int tap = kk >> cinShift;
    const int c0  = kk & cmask;
    const int ky  = tap / ksz;
    const int kx  = tap - ky * ksz;

    __syncthreads();
    // ---- B tile: 32 positions x 32 cin, 16B chunks (32*4 = 128 uint4) ----
    for (int q = tid; q < 128; q += nthr) {
      int p  = q >> 2;
      int cc = (q & 3) << 3;           // f16 offset within the 32-cin slab
      int yi = pos_y[p] * stride + ky - pad;
      int xi = pos_x[p] * stride + kx - pad;
      uint4 v = {0u, 0u, 0u, 0u};
      if (yi >= 0 && yi < Hi && xi >= 0 && xi < Wi) {
        const _Float16* src = X + ((((size_t)pos_b[p] * Hi + yi) * Wi + xi) * Cin + c0 + cc);
        v = *(const uint4*)src;
      }
      *(uint4*)(&b_s[p][cc]) = v;
    }
    // ---- A tile: 32 k-rows x rowW couts, 16B chunks (32 * rowW/8) ----
    const int chunksA = 32 << rsh;
    for (int q = tid; q < chunksA; q += nthr) {
      int kr = q >> rsh;
      int cc = (q & ((1 << rsh) - 1)) << 3;
      const _Float16* src = Wt + ((size_t)(kk + kr) * Cout + m_base + cc);
      *(uint4*)(&a_s[kr][cc]) = *(const uint4*)src;
    }
    __syncthreads();

    // ---- gather fragments per ISA 16-bit A(16x32)/B(32x16) layouts ----
    union { v16h v; _Float16 h[16]; } af, bf0, bf1;
#pragma unroll
    for (int e = 0; e < 16; ++e) {
      int ka = hsel ? ((e < 8) ? (e + 8) : (e + 16))   // lanes 16-31: K 8..15, 24..31
                    : ((e < 8) ? e : (e + 8));          // lanes 0-15 : K 0..7, 16..23
      af.h[e]  = a_s[ka][wave * 16 + row];
      bf0.h[e] = b_s[row][ka];                          // cols = positions 0..15
      bf1.h[e] = b_s[16 + row][ka];                     // cols = positions 16..31
    }
    acc0.v = __builtin_amdgcn_wmma_f32_16x16x32_f16(false, af.v, false, bf0.v,
                                                    (short)0, acc0.v, false, false);
    acc1.v = __builtin_amdgcn_wmma_f32_16x16x32_f16(false, af.v, false, bf1.v,
                                                    (short)0, acc1.v, false, false);
  }

  // ---- epilogue: C/D layout col=lane&15, vgpr r -> M = r + hsel*8; 16B stores ----
#pragma unroll
  for (int t = 0; t < 2; ++t) {
    const int colp = t * 16 + row;
    const size_t obase =
        (((size_t)pos_b[colp] * Ho + pos_y[colp]) * Wo + pos_x[colp]) * (size_t)Cout;
    const float* af_ = t ? acc1.f : acc0.f;
    union { uint4 u; _Float16 h[8]; } rv, ov;
    if (resid) rv.u = *(const uint4*)(resid + obase + mstart);
#pragma unroll
    for (int r = 0; r < 8; ++r) {
      float v = af_[r] * scl[r] + sft[r];
      if (resid) v += (float)rv.h[r];
      if (do_relu) v = fmaxf(v, 0.f);
      ov.h[r] = (_Float16)v;
    }
    *(uint4*)(Y + obase + mstart) = ov.u;
  }
}

// ---------------- PrimaryCaps LayerNorm: NHWC u -> xcaps [B,336,16] f32 ----------------
__global__ void primarycaps_ln_kernel(const _Float16* __restrict__ U,
                                      const float* __restrict__ g,
                                      const float* __restrict__ be,
                                      float* __restrict__ xcaps,
                                      int B, int Hc, int Wc, int cap_off, int caps_total) {
  int idx = blockIdx.x * blockDim.x + threadIdx.x;
  int total = B * 4 * Hc * Wc;
  if (idx >= total) return;
  int w = idx % Wc; int t = idx / Wc;
  int h = t % Hc; t /= Hc;
  int gi = t % 4; int b = t / 4;
  const _Float16* src = U + ((((size_t)b * Hc + h) * Wc + w) * 64 + gi * 16);
  float v[16]; float mu = 0.f;
#pragma unroll
  for (int d = 0; d < 16; ++d) { v[d] = (float)src[d]; mu += v[d]; }
  mu *= (1.f / 16.f);
  float var = 0.f;
#pragma unroll
  for (int d = 0; d < 16; ++d) { float z = v[d] - mu; var += z * z; }
  var *= (1.f / 16.f);
  float inv = rsqrtf(var + EPSLN);
  float* dst = xcaps + ((size_t)b * caps_total + cap_off + (gi * Hc * Wc + h * Wc + w)) * 16;
#pragma unroll
  for (int d = 0; d < 16; ++d) dst[d] = (v[d] - mu) * inv * g[d] + be[d];
}

// ---------------- routing: softmax row-sums are identically 1 -> c = 1 + rowsum(bias) ----------------
__global__ void bias_rowsum_kernel(const float* __restrict__ bias, float* __restrict__ crow,
                                   int rows, int ncols) {
  int r = blockIdx.x * blockDim.x + threadIdx.x;
  if (r >= rows) return;
  const float* p = bias + (size_t)r * ncols;
  float s = 0.f;
  for (int j = 0; j < ncols; ++j) s += p[j];
  crow[r] = s;
}

// S[b,k,l] = sum_i (1+crow[k,i]) * sum_j x[b,i,j] * W[k,i,j,l]
__global__ void routing_US_kernel(const float* __restrict__ x, const float* __restrict__ W,
                                  const float* __restrict__ crow, float* __restrict__ S,
                                  int B, int N0, int D0, int N1, int D1) {
  int idx = blockIdx.x * blockDim.x + threadIdx.x;
  int total = B * N1 * D1;
  if (idx >= total) return;
  int l = idx % D1; int t = idx / D1;
  int k = t % N1;  int b = t / N1;
  float s = 0.f;
  for (int i = 0; i < N0; ++i) {
    const float* xi = x + ((size_t)b * N0 + i) * D0;
    const float* wi = W + (((size_t)k * N0 + i) * D0) * D1 + l;
    float dot = 0.f;
    for (int j = 0; j < D0; ++j) dot += xi[j] * wi[(size_t)j * D1];
    s += (1.f + crow[k * N0 + i]) * dot;
  }
  S[idx] = s;
}

__global__ void squash_kernel(const float* __restrict__ S, float* __restrict__ out,
                              int B, int N1, int D1) {
  int idx = blockIdx.x * blockDim.x + threadIdx.x;
  int total = B * N1;
  if (idx >= total) return;
  const float* p = S + (size_t)idx * D1;
  float sn = 0.f;
  for (int l = 0; l < D1; ++l) sn += p[l] * p[l];
  float f = sn / (1.f + sn) / (sqrtf(sn) + EPSSQ);
  float* o = out + (size_t)idx * D1;
  for (int l = 0; l < D1; ++l) o[l] = p[l] * f;
}

// ---------------- host orchestration ----------------
extern "C" void kernel_launch(void* const* d_in, const int* in_sizes, int n_in,
                              void* d_out, int out_size, void* d_ws, size_t ws_size,
                              hipStream_t stream) {
  (void)in_sizes; (void)n_in; (void)ws_size; (void)out_size;
  const float* x = (const float*)d_in[0];

  size_t off = 0;
  auto carve = [&](size_t bytes) -> void* {
    void* p = (char*)d_ws + off;
    off += (bytes + 255) & ~(size_t)255;
    return p;
  };
  _Float16* act1 = (_Float16*)carve((size_t)8 * 32 * 32 * 64  * sizeof(_Float16));
  _Float16* act2 = (_Float16*)carve((size_t)8 * 32 * 32 * 128 * sizeof(_Float16));
  _Float16* act3 = (_Float16*)carve((size_t)8 * 32 * 32 * 128 * sizeof(_Float16));
  _Float16* act4 = (_Float16*)carve((size_t)8 * 32 * 32 * 256 * sizeof(_Float16));
  _Float16* actu = (_Float16*)carve((size_t)8 * 8 * 8 * 64    * sizeof(_Float16));
  _Float16* wt   = (_Float16*)carve((size_t)1152 * 256        * sizeof(_Float16));
  float* scale = (float*)carve(256 * sizeof(float));
  float* shift = (float*)carve(256 * sizeof(float));
  float* xcaps = (float*)carve((size_t)8 * 336 * 16 * sizeof(float));
  float* crow1 = (float*)carve((size_t)168 * 336 * sizeof(float));
  float* crow2 = (float*)carve((size_t)10 * 168 * sizeof(float));
  float* S1    = (float*)carve((size_t)8 * 168 * 24 * sizeof(float));
  float* x2    = (float*)carve((size_t)8 * 168 * 24 * sizeof(float));
  float* S2    = (float*)carve((size_t)8 * 10 * 32 * sizeof(float));

  const int strides[3] = {1, 2, 4};
  for (int br = 0; br < 3; ++br) {
    int s = strides[br];
    int H1 = 32 / s;

    { // conv1: 3->64 k5 p2 stride s, BN+ReLU
      int base = 1 + (br * 4 + 0) * 6;
      const float* w  = (const float*)d_in[base + 0];
      const float* b  = (const float*)d_in[base + 1];
      const float* g  = (const float*)d_in[base + 2];
      const float* be = (const float*)d_in[base + 3];
      const float* mu = (const float*)d_in[base + 4];
      const float* vr = (const float*)d_in[base + 5];
      prep_ss_kernel<<<1, 64, 0, stream>>>(b, g, be, mu, vr, scale, shift, 64, 1);
      int total = 8 * H1 * H1 * 64;
      conv1_kernel<<<(total + 255) / 256, 256, 0, stream>>>(x, w, scale, shift, act1,
                                                            8, 32, 32, H1, H1, s);
    }

    struct { int cin, cshift, cout; _Float16 *in, *out, *res; } L[3] = {
      { 64, 6, 128, act1, act2, nullptr },
      { 128, 7, 128, act2, act3, act2   },   // residual (channels match), pre-ReLU add
      { 128, 7, 256, act3, act4, nullptr },
    };
    for (int ly = 0; ly < 3; ++ly) {
      int base = 1 + (br * 4 + (ly + 1)) * 6;
      const float* w  = (const float*)d_in[base + 0];
      const float* b  = (const float*)d_in[base + 1];
      const float* g  = (const float*)d_in[base + 2];
      const float* be = (const float*)d_in[base + 3];
      const float* mu = (const float*)d_in[base + 4];
      const float* vr = (const float*)d_in[base + 5];
      int Cin = L[ly].cin, Cout = L[ly].cout;
      prep_ss_kernel<<<1, 256, 0, stream>>>(b, g, be, mu, vr, scale, shift, Cout, 1);
      int wtotal = Cout * Cin * 9;
      prep_wt_kernel<<<(wtotal + 255) / 256, 256, 0, stream>>>(w, wt, Cout, Cin, 3);
      int N = 8 * H1 * H1;                       // multiple of 32 for all branches
      dim3 grid(N / 32, Cout / 128);             // 8 waves -> 128 couts per block
      conv_wmma_kernel<<<grid, 256, 0, stream>>>(L[ly].in, wt, scale, shift, L[ly].res,
                                                 L[ly].out, 8, H1, H1, Cin, L[ly].cshift,
                                                 Cout, H1, H1, 3, 1, 1, 1);
    }

    { // PrimaryCaps: 256->64, k4 s4 p0, bias only, no relu; then LayerNorm over caps_dim=16
      int pb = 73 + br * 4;
      const float* w  = (const float*)d_in[pb + 0];
      const float* b  = (const float*)d_in[pb + 1];
      const float* g  = (const float*)d_in[pb + 2];
      const float* be = (const float*)d_in[pb + 3];
      prep_ss_kernel<<<1, 64, 0, stream>>>(b, nullptr, nullptr, nullptr, nullptr,
                                           scale, shift, 64, 0);
      int wtotal = 64 * 256 * 16;
      prep_wt_kernel<<<(wtotal + 255) / 256, 256, 0, stream>>>(w, wt, 64, 256, 4);
      int Hc = H1 / 4;
      int N = 8 * Hc * Hc;                       // 512 / 128 / 32, all multiples of 32
      dim3 grid(N / 32, 1);                      // 4 waves -> 64 couts per block
      conv_wmma_kernel<<<grid, 128, 0, stream>>>(act4, wt, scale, shift, nullptr, actu,
                                                 8, H1, H1, 256, 8, 64, Hc, Hc, 4, 4, 0, 0);
      int cap_off = (br == 0) ? 0 : ((br == 1) ? 256 : 320);
      int ctotal = 8 * 4 * Hc * Hc;
      primarycaps_ln_kernel<<<(ctotal + 127) / 128, 128, 0, stream>>>(
          actu, g, be, xcaps, 8, Hc, Hc, cap_off, 336);
    }
  }

  { // routing 1: (336,16) -> (168,24); softmax rows sum to 1 => c = 1 + rowsum(b1)
    const float* W1 = (const float*)d_in[85];
    const float* b1 = (const float*)d_in[86];
    int rows = 168 * 336;
    bias_rowsum_kernel<<<(rows + 255) / 256, 256, 0, stream>>>(b1, crow1, rows, 336);
    int total = 8 * 168 * 24;
    routing_US_kernel<<<(total + 255) / 256, 256, 0, stream>>>(xcaps, W1, crow1, S1,
                                                               8, 336, 16, 168, 24);
    int bk = 8 * 168;
    squash_kernel<<<(bk + 255) / 256, 256, 0, stream>>>(S1, x2, 8, 168, 24);
  }
  { // routing 2: (168,24) -> (10,32)
    const float* W2 = (const float*)d_in[87];
    const float* b2 = (const float*)d_in[88];
    int rows = 10 * 168;
    bias_rowsum_kernel<<<(rows + 255) / 256, 256, 0, stream>>>(b2, crow2, rows, 168);
    int total = 8 * 10 * 32;
    routing_US_kernel<<<(total + 255) / 256, 256, 0, stream>>>(x2, W2, crow2, S2,
                                                               8, 168, 24, 10, 32);
    int bk = 8 * 10;
    squash_kernel<<<(bk + 255) / 256, 256, 0, stream>>>(S2, (float*)d_out, 8, 10, 32);
  }
}
